// Head_33397665694114
// MI455X (gfx1250) — compile-verified, matrix operand order checked by reading, not run on hardware
//
#include <hip/hip_runtime.h>
#include <hip/hip_bf16.h>

// Problem constants (match reference)
constexpr int NB = 8;     // batch
constexpr int NT = 2048;  // sequence length
constexpr int NC = 1024;  // n_embd
constexpr int NH = 64;    // head size

typedef __attribute__((ext_vector_type(16))) _Float16 v16h;
typedef __attribute__((ext_vector_type(8)))  float    v8f;

union V16H {
  v16h v;
  _Float16 h[16];
  unsigned u[8];
};

#if defined(__has_builtin)
#if __has_builtin(__builtin_amdgcn_sched_barrier)
#define SCHED_FENCE() __builtin_amdgcn_sched_barrier(0)
#endif
#endif
#ifndef SCHED_FENCE
#define SCHED_FENCE() asm volatile("" ::: "memory")
#endif

// K offset of the packed f16 pair held in "VGPR" i by this lane-half
// (16-bit A/B 16x32 WMMA layout):
//   lanes 0-15 : vgpr0..3 -> K={0,1},{2,3},{4,5},{6,7}; vgpr4..7 -> K={16..23}
//   lanes 16-31: +8 on each block.
__device__ __forceinline__ int kpair(int i, int hl) {
  return ((i < 4) ? (2 * i) : (16 + 2 * (i - 4))) + 8 * hl;
}

// ---- raw f32 tile: the 16 floats this lane contributes to one 16x32 operand ----
struct Raw16 {
  float4 a, b, c, d;  // floats [off..off+3],[off+4..off+7],[off+16..off+19],[off+20..off+23]
};

__device__ __forceinline__ Raw16 load_raw_f32(const float* __restrict__ base, int ldr,
                                              int ml, int hl, int kbase) {
  const float* p = base + (size_t)ml * ldr + kbase + 8 * hl;
  Raw16 r;
  r.a = *(const float4*)(p);
  r.b = *(const float4*)(p + 4);
  r.c = *(const float4*)(p + 16);
  r.d = *(const float4*)(p + 20);
  return r;
}

__device__ __forceinline__ v16h cvt_tile(const Raw16& r) {
  V16H t;
  t.h[0]  = (_Float16)r.a.x;  t.h[1]  = (_Float16)r.a.y;
  t.h[2]  = (_Float16)r.a.z;  t.h[3]  = (_Float16)r.a.w;
  t.h[4]  = (_Float16)r.b.x;  t.h[5]  = (_Float16)r.b.y;
  t.h[6]  = (_Float16)r.b.z;  t.h[7]  = (_Float16)r.b.w;
  t.h[8]  = (_Float16)r.c.x;  t.h[9]  = (_Float16)r.c.y;
  t.h[10] = (_Float16)r.c.z;  t.h[11] = (_Float16)r.c.w;
  t.h[12] = (_Float16)r.d.x;  t.h[13] = (_Float16)r.d.y;
  t.h[14] = (_Float16)r.d.z;  t.h[15] = (_Float16)r.d.w;
  return t.v;
}

// f16-source operand tile load (packed u32 pairs; k offsets always even).
__device__ __forceinline__ v16h load_tile_f16(const _Float16* __restrict__ base, int ldr,
                                              int ml, int hl, int kbase) {
  const _Float16* p = base + (size_t)ml * ldr + kbase;
  V16H t;
#pragma unroll
  for (int i = 0; i < 8; ++i) {
    int k0 = kpair(i, hl);
    t.u[i] = *(const unsigned*)(p + k0);
  }
  return t.v;
}

__device__ __forceinline__ v8f wmma_f16(v16h a, v16h b, v8f c) {
  return __builtin_amdgcn_wmma_f32_16x16x32_f16(false, a, false, b, (short)0, c,
                                                false, false);
}

// ---------------------------------------------------------------------------
// Kernel 1: q/k/v projections.  Each wave: one 16-row x 16-col tile of all 3.
//   q,k stored [B*T, H] f16 row-major; v stored transposed [B, H, T] f16.
// ---------------------------------------------------------------------------
__global__ __launch_bounds__(256) void proj_kernel(
    const float* __restrict__ x, const float* __restrict__ Wk,
    const float* __restrict__ Wq, const float* __restrict__ Wv,
    _Float16* __restrict__ qb, _Float16* __restrict__ kb,
    _Float16* __restrict__ vtb) {
  const int lane = threadIdx.x & 31;
  const int wave = blockIdx.x * (blockDim.x >> 5) + (threadIdx.x >> 5);
  const int rt = wave >> 2;   // 0..(B*T/16 - 1)
  const int nt = wave & 3;    // 0..3 (H/16)
  const int hl = lane >> 4;
  const int ml = lane & 15;

  const int row0 = rt * 16;
  const int h0 = nt * 16;

  v8f aq = {}; v8f ak = {}; v8f av = {};
  const float* xrow  = x  + (size_t)row0 * NC;
  const float* wqrow = Wq + (size_t)h0 * NC;
  const float* wkrow = Wk + (size_t)h0 * NC;
  const float* wvrow = Wv + (size_t)h0 * NC;

  for (int kbase = 0; kbase < NC; kbase += 32) {
    // Phase 1: issue ALL 16 global b128 loads for this k-step.
    Raw16 rx = load_raw_f32(xrow,  NC, ml, hl, kbase);
    Raw16 rq = load_raw_f32(wqrow, NC, ml, hl, kbase);
    Raw16 rk = load_raw_f32(wkrow, NC, ml, hl, kbase);
    Raw16 rv = load_raw_f32(wvrow, NC, ml, hl, kbase);
    // Hard scheduling fence: nothing may cross, so loads stay grouped and the
    // backend cannot collapse them into a load->wait->convert serial chain.
    SCHED_FENCE();

    // Phase 2: convert to f16 operand tiles.
    v16h a  = cvt_tile(rx);
    v16h bq = cvt_tile(rq);
    v16h bk = cvt_tile(rk);
    v16h bv = cvt_tile(rv);

    // Phase 3: matrix ops.
    aq = wmma_f16(a, bq, aq);
    ak = wmma_f16(a, bk, ak);
    av = wmma_f16(a, bv, av);
  }

#pragma unroll
  for (int r = 0; r < 8; ++r) {
    const int row = row0 + r + 8 * hl;   // flat b*T + t
    const int h = h0 + ml;
    qb[(size_t)row * NH + h] = (_Float16)aq[r];
    kb[(size_t)row * NH + h] = (_Float16)ak[r];
    const int bidx = row >> 11;          // / NT
    const int t = row & (NT - 1);
    vtb[((size_t)bidx * NH + h) * NT + t] = (_Float16)av[r];
  }
}

// ---------------------------------------------------------------------------
// Kernel 2: flash attention. Each wave: 16 query rows, key chunks of 32.
// Softmax runs in the exp2 domain: log2(e) is folded into the score scale so
// every exponential is a single native v_exp_f32.
// ---------------------------------------------------------------------------
__global__ __launch_bounds__(256) void attn_kernel(
    const _Float16* __restrict__ qb, const _Float16* __restrict__ kb,
    const _Float16* __restrict__ vtb, float* __restrict__ out) {
  __shared__ _Float16 pbuf[8][16][36];   // padded rows -> conflict-free A-layout reads

  const int lane = threadIdx.x & 31;
  const int wslot = threadIdx.x >> 5;
  const int wave = blockIdx.x * 8 + wslot;
  const int b = wave >> 7;      // / (T/16)
  const int qt = wave & 127;
  const int q0 = qt * 16;
  const int hl = lane >> 4;
  const int ml = lane & 15;

  // Q tile (A operand), K-dim = H = 64 -> two k-steps
  const _Float16* qrow = qb + ((size_t)b * NT + q0) * NH;
  const v16h aq0 = load_tile_f16(qrow, NH, ml, hl, 0);
  const v16h aq1 = load_tile_f16(qrow, NH, ml, hl, 32);

  v8f o0 = {}, o1 = {}, o2 = {}, o3 = {};
  float m[8], l[8];
#pragma unroll
  for (int r = 0; r < 8; ++r) { m[r] = -__builtin_inff(); l[r] = 0.f; }

  // (1/sqrt(n_embd)) * log2(e): softmax computed with exp2.
  const float sscale2 = 0.03125f * 1.4426950408889634f;
  const int nchunks = (q0 + 16 + 31) >> 5;
  const _Float16* kb_b  = kb + (size_t)b * NT * NH;
  const _Float16* vtb_b = vtb + (size_t)b * NH * NT;

  for (int c = 0; c < nchunks; ++c) {
    const int s0 = c * 32;

    // ---- issue ALL K and V tile loads up front; V latency hides under
    //      the S-WMMAs + softmax VALU work ----
    const _Float16* krow0 = kb_b + (size_t)s0 * NH;
    const _Float16* krow1 = krow0 + (size_t)16 * NH;
    v16h bk00 = load_tile_f16(krow0, NH, ml, hl, 0);
    v16h bk01 = load_tile_f16(krow0, NH, ml, hl, 32);
    v16h bk10 = load_tile_f16(krow1, NH, ml, hl, 0);
    v16h bk11 = load_tile_f16(krow1, NH, ml, hl, 32);
    v16h bv0 = load_tile_f16(vtb_b + (size_t)0  * NT, NT, ml, hl, s0);
    v16h bv1 = load_tile_f16(vtb_b + (size_t)16 * NT, NT, ml, hl, s0);
    v16h bv2 = load_tile_f16(vtb_b + (size_t)32 * NT, NT, ml, hl, s0);
    v16h bv3 = load_tile_f16(vtb_b + (size_t)48 * NT, NT, ml, hl, s0);
    SCHED_FENCE();

    // ---- S = Q K^T  (two 16x16 f32 tiles over a 32-key chunk) ----
    v8f s0acc = {}, s1acc = {};
    s0acc = wmma_f16(aq0, bk00, s0acc);
    s0acc = wmma_f16(aq1, bk01, s0acc);
    s1acc = wmma_f16(aq0, bk10, s1acc);
    s1acc = wmma_f16(aq1, bk11, s1acc);

    // ---- scale + causal mask + row max (reduce across 16-lane half) ----
    float mx[8];
#pragma unroll
    for (int r = 0; r < 8; ++r) {
      const int trow = q0 + r + 8 * hl;
      float v0 = s0acc[r] * sscale2;
      float v1 = s1acc[r] * sscale2;
      if (s0 + ml > trow)      v0 = -__builtin_inff();
      if (s0 + 16 + ml > trow) v1 = -__builtin_inff();
      s0acc[r] = v0; s1acc[r] = v1;
      float t = fmaxf(v0, v1);
      t = fmaxf(t, __shfl_xor(t, 1, 32));
      t = fmaxf(t, __shfl_xor(t, 2, 32));
      t = fmaxf(t, __shfl_xor(t, 4, 32));
      t = fmaxf(t, __shfl_xor(t, 8, 32));
      mx[r] = t;
    }

    // ---- online softmax (exp2 domain); stage P into LDS (16x32 row-major) ----
#pragma unroll
    for (int r = 0; r < 8; ++r) {
      const float newm = fmaxf(m[r], mx[r]);
      const float corr = __builtin_amdgcn_exp2f(m[r] - newm);
      m[r] = newm;
      const float p0 = __builtin_amdgcn_exp2f(s0acc[r] - newm);
      const float p1 = __builtin_amdgcn_exp2f(s1acc[r] - newm);
      float rs = p0 + p1;
      rs += __shfl_xor(rs, 1, 32);
      rs += __shfl_xor(rs, 2, 32);
      rs += __shfl_xor(rs, 4, 32);
      rs += __shfl_xor(rs, 8, 32);
      l[r] = l[r] * corr + rs;
      o0[r] *= corr; o1[r] *= corr; o2[r] *= corr; o3[r] *= corr;
      pbuf[wslot][r + 8 * hl][ml]      = (_Float16)p0;
      pbuf[wslot][r + 8 * hl][16 + ml] = (_Float16)p1;
    }

    // Same-wave LDS is in-order; drain DScnt before re-reading our slot.
    asm volatile("s_wait_dscnt 0" ::: "memory");

    // ---- reload P in A layout (16x32, K = keys) ----
    V16H ap;
#pragma unroll
    for (int i = 0; i < 8; ++i) {
      const int k0 = kpair(i, hl);
      ap.u[i] = *(const unsigned*)&pbuf[wslot][ml][k0];
    }

    // ---- O += P V  (V tiles already resident) ----
    o0 = wmma_f16(ap.v, bv0, o0);
    o1 = wmma_f16(ap.v, bv1, o1);
    o2 = wmma_f16(ap.v, bv2, o2);
    o3 = wmma_f16(ap.v, bv3, o3);
  }

  // ---- normalize and store f32 output [B, T, H] ----
#pragma unroll
  for (int r = 0; r < 8; ++r) {
    const float inv = 1.0f / l[r];
    const int row = q0 + r + 8 * hl;
    const size_t base = ((size_t)b * NT + row) * NH;
    out[base + 0  + ml] = o0[r] * inv;
    out[base + 16 + ml] = o1[r] * inv;
    out[base + 32 + ml] = o2[r] * inv;
    out[base + 48 + ml] = o3[r] * inv;
  }
}

// ---------------------------------------------------------------------------
extern "C" void kernel_launch(void* const* d_in, const int* in_sizes, int n_in,
                              void* d_out, int out_size, void* d_ws, size_t ws_size,
                              hipStream_t stream) {
  const float* x  = (const float*)d_in[0];
  const float* Wk = (const float*)d_in[1];
  const float* Wq = (const float*)d_in[2];
  const float* Wv = (const float*)d_in[3];

  const size_t nqkv = (size_t)NB * NT * NH;  // 1M elems each
  _Float16* qb  = (_Float16*)d_ws;
  _Float16* kbf = qb + nqkv;
  _Float16* vtb = kbf + nqkv;

  // Projection: B*T/16 row-tiles * 4 n-tiles = 4096 waves; 8 waves/block.
  proj_kernel<<<512, 256, 0, stream>>>(x, Wk, Wq, Wv, qb, kbf, vtb);

  // Attention: B * T/16 = 1024 waves; 8 waves/block.
  attn_kernel<<<128, 256, 0, stream>>>(qb, kbf, vtb, (float*)d_out);
}
